// Model_85564338471455
// MI455X (gfx1250) — matrix operand order repaired and checked by reading, compiled, not run on hardware
//
#include <hip/hip_runtime.h>
#include <hip/hip_bf16.h>
#include <math.h>

#define CH 62
#define HID 4096
#define NEG_SLOPE 0.2f

typedef unsigned int u32;
typedef unsigned short u16;
typedef __attribute__((ext_vector_type(16))) __bf16 v16bf;
typedef __attribute__((ext_vector_type(8)))  float  v8f;
typedef __attribute__((ext_vector_type(4)))  u32    u32x4;

__device__ __forceinline__ u16 bf_bits(float f) {
  union { __bf16 b; u16 u; } v;
  v.b = (__bf16)f;             // native f32->bf16 convert
  return v.u;
}

// ---------------- fp32 -> bf16 pack (activations only; weights converted in-GEMM)
__global__ void pack_bf16(const float* __restrict__ in, u16* __restrict__ out, int total) {
  int i = blockIdx.x * blockDim.x + threadIdx.x;
  if (i < total) out[i] = bf_bits(in[i]);
}

// ===================== WMMA GEMM =====================
// C[64,HID] = A_bf16[64,K] x W_fp32[K,HID]  (bf16 WMMA, fp32 acc)
// 128 threads = 4 waves; each wave owns a 64x16 strip of C: 4 M-tiles share one
// B (weight) fragment, so each weight element is streamed from HBM exactly once.
// K-chunks are double-buffered in registers: loads for chunk k+32 are issued
// before the WMMAs of chunk k, hiding memory latency behind compute.
// All accesses are SGPR-base + 32-bit VGPR element offset with constant
// instruction offsets (N = HID is compile-time), so the hot loop has no 64-bit
// address arithmetic and emits pure global_load saddr forms.
// Requires K % 64 == 0 (holds: 1280, 4096).

struct KChunk {
  u32x4 a0[4];   // A fragment VGPR0-3 per M-tile (K a_k0 .. a_k0+7)
  u32x4 a1[4];   // A fragment VGPR4-7 per M-tile (K a_k0+16 .. a_k0+23)
  float w[16];   // raw fp32 weights, K = b_k0 .. b_k0+7, b_k0+16 .. b_k0+23 for this lane's column
};

__device__ __forceinline__ void load_chunk(KChunk& c, const __bf16* __restrict__ A,
                                           const u32 aoff[4],
                                           const float* __restrict__ W, u32 woff,
                                           int koff) {
  #pragma unroll
  for (int m = 0; m < 4; ++m) {
    c.a0[m] = *(const u32x4*)(A + aoff[m] + koff);
    c.a1[m] = *(const u32x4*)(A + aoff[m] + koff + 16);
  }
  #pragma unroll
  for (int j = 0; j < 16; ++j)
    c.w[j] = W[woff + (u32)(koff * HID) + (u32)(((j & 7) + (j >> 3) * 16) * HID)];   // imm offsets
}

__device__ __forceinline__ void compute_chunk(const KChunk& c, v8f acc[4]) {
  v16bf bv;
  #pragma unroll
  for (int j = 0; j < 16; ++j) bv[j] = (__bf16)c.w[j];   // v_cvt_pk_bf16_f32
  #pragma unroll
  for (int m = 0; m < 4; ++m) {
    union { v16bf v; u32x4 q[2]; } af;
    af.q[0] = c.a0[m];
    af.q[1] = c.a1[m];
    acc[m] = __builtin_amdgcn_wmma_f32_16x16x32_bf16(
        false, af.v, false, bv, (short)0, acc[m], false, false);
  }
}

__global__ __launch_bounds__(128, 1)
void gat_gemm(const __bf16* __restrict__ A, const float* __restrict__ W,
              float* __restrict__ C, int K) {
  const int lane   = threadIdx.x & 31;
  const int wave   = threadIdx.x >> 5;
  const int n0     = (blockIdx.x * 4 + wave) * 16;
  const int col    = n0 + (lane & 15);
  const int halfHi = lane >> 4;        // 0: lanes 0-15, 1: lanes 16-31
  const int a_k0   = halfHi * 8;       // A half-wave K offset
  const int b_k0   = halfHi * 8;       // B half-wave K offset

  v8f z = {0.f,0.f,0.f,0.f,0.f,0.f,0.f,0.f};
  v8f acc[4] = {z, z, z, z};

  // 32-bit element offsets, advanced per 64-wide K step
  u32 aoff[4];
  #pragma unroll
  for (int m = 0; m < 4; ++m)
    aoff[m] = (u32)(m * 16 + (lane & 15)) * (u32)K + (u32)a_k0;
  u32 woff = (u32)b_k0 * HID + (u32)col;

  KChunk ca, cb;
  load_chunk(ca, A, aoff, W, woff, 0);

  for (int kb = 0; kb < K; kb += 64) {
    load_chunk(cb, A, aoff, W, woff, 32);
    compute_chunk(ca, acc);                 // waits on loads issued last round
    #pragma unroll
    for (int m = 0; m < 4; ++m) aoff[m] += 64;
    woff += 64 * HID;
    if (kb + 64 < K)
      load_chunk(ca, A, aoff, W, woff, 0);
    compute_chunk(cb, acc);
  }

  const u32 cbase = (u32)(halfHi * 8) * HID + (u32)col;
  #pragma unroll
  for (int m = 0; m < 4; ++m) {
    #pragma unroll
    for (int r = 0; r < 8; ++r) {
      // C layout: VGPR r -> M = m*16 + r (lanes 0-15) / m*16 + 8 + r (lanes 16-31)
      C[cbase + (u32)((m * 16 + r) * HID)] = acc[m][r];
    }
  }
}

// ---------------- s[i] = h[i]·att_src, d[i] = h[i]·att_dst
__global__ void row_dots(const float* __restrict__ g, const float* __restrict__ as,
                         const float* __restrict__ ad, float* __restrict__ s,
                         float* __restrict__ d, int N) {
  __shared__ float bs[256], bd[256];
  int row = blockIdx.x;
  float a0 = 0.f, a1 = 0.f;
  for (int i = threadIdx.x; i < N; i += 256) {
    float h = g[(size_t)row * N + i];
    a0 = fmaf(h, as[i], a0);
    a1 = fmaf(h, ad[i], a1);
  }
  bs[threadIdx.x] = a0; bd[threadIdx.x] = a1;
  __syncthreads();
  for (int st = 128; st > 0; st >>= 1) {
    if (threadIdx.x < st) {
      bs[threadIdx.x] += bs[threadIdx.x + st];
      bd[threadIdx.x] += bd[threadIdx.x + st];
    }
    __syncthreads();
  }
  if (threadIdx.x == 0) { s[row] = bs[0]; d[row] = bd[0]; }
}

// ---------------- bucket edges (+ self loops) by destination (counting sort in LDS)
__global__ void build_csr(const int* __restrict__ ei, int E, int* __restrict__ offs,
                          int* __restrict__ srcs) {
  __shared__ int cnt[CH];
  __shared__ int cur[CH];
  int tid = threadIdx.x;
  if (tid < CH) cnt[tid] = 0;
  __syncthreads();
  int Etot = E + CH;
  for (int i = tid; i < Etot; i += blockDim.x) {
    int dst = (i < E) ? ei[E + i] : (i - E);
    atomicAdd(&cnt[dst], 1);
  }
  __syncthreads();
  if (tid == 0) {
    int run = 0;
    for (int c = 0; c < CH; ++c) { cur[c] = run; offs[c] = run; run += cnt[c]; }
    offs[CH] = run;
  }
  __syncthreads();
  for (int i = tid; i < Etot; i += blockDim.x) {
    int dst = (i < E) ? ei[E + i] : (i - E);
    int src = (i < E) ? ei[i]     : (i - E);
    int pos = atomicAdd(&cur[dst], 1);
    srcs[pos] = src;
  }
}

// ---------------- per-destination edge softmax: leaky_relu -> max -> exp/sum -> norm
__global__ void edge_softmax(const float* __restrict__ s, const float* __restrict__ dv,
                             const int* __restrict__ offs, const int* __restrict__ srcs,
                             float* __restrict__ alpha) {
  __shared__ float red[64];
  int dst = blockIdx.x, tid = threadIdx.x;
  int beg = offs[dst], end = offs[dst + 1];
  float dd = dv[dst];
  float lm = -3.0e38f;
  for (int i = beg + tid; i < end; i += 64) {
    float e = s[srcs[i]] + dd;
    e = (e > 0.f) ? e : NEG_SLOPE * e;
    alpha[i] = e;
    lm = fmaxf(lm, e);
  }
  red[tid] = lm; __syncthreads();
  for (int st = 32; st > 0; st >>= 1) {
    if (tid < st) red[tid] = fmaxf(red[tid], red[tid + st]);
    __syncthreads();
  }
  float m = red[0];
  __syncthreads();
  float ls = 0.f;
  for (int i = beg + tid; i < end; i += 64) {
    float w = expf(alpha[i] - m);
    alpha[i] = w;
    ls += w;
  }
  red[tid] = ls; __syncthreads();
  for (int st = 32; st > 0; st >>= 1) {
    if (tid < st) red[tid] += red[tid + st];
    __syncthreads();
  }
  float inv = 1.0f / red[0];
  for (int i = beg + tid; i < end; i += 64) alpha[i] *= inv;
}

// ---------------- out[dst] = sum_e alpha_e * h[src_e] + bias ; optional relu
// writes fp32 (for final concat) AND bf16-packed A matrix for the next WMMA GEMM
__global__ void aggregate(const float* __restrict__ g, const float* __restrict__ alpha,
                          const int* __restrict__ offs, const int* __restrict__ srcs,
                          const float* __restrict__ bias, float* __restrict__ out32,
                          u16* __restrict__ outbf, int N, int relu) {
  int dst = blockIdx.y;
  int f = blockIdx.x * 256 + threadIdx.x;
  int beg = offs[dst], end = offs[dst + 1];
  float acc = 0.f;
  for (int i = beg; i < end; ++i)
    acc = fmaf(alpha[i], g[(size_t)srcs[i] * N + f], acc);
  acc += bias[f];
  if (relu) acc = fmaxf(acc, 0.f);
  out32[(size_t)dst * N + f] = acc;
  outbf[(size_t)dst * N + f] = bf_bits(acc);
}

// ---------------- hd = relu(demo @ fcW + fcb)   (3 -> 4096)
__global__ void fc_demo(const float* __restrict__ demo, const float* __restrict__ fcW,
                        const float* __restrict__ fcb, float* __restrict__ hd) {
  int f = blockIdx.x * 256 + threadIdx.x;
  float acc = fcb[f];
  acc = fmaf(demo[0], fcW[f], acc);
  acc = fmaf(demo[1], fcW[HID + f], acc);
  acc = fmaf(demo[2], fcW[2 * HID + f], acc);
  hd[f] = fmaxf(acc, 0.f);
}

// ---------------- final projection [1,512000] @ [512000,2]: stage 1 partials
__global__ void final_partial(const float* __restrict__ hf, const float* __restrict__ ht,
                              const float* __restrict__ hd, const float* __restrict__ oW,
                              float* __restrict__ part, int L) {
  __shared__ float r0[256], r1[256];
  float a0 = 0.f, a1 = 0.f;
  for (int i = blockIdx.x * 256 + threadIdx.x; i < L; i += gridDim.x * 256) {
    float v;
    if (i < CH * HID)          v = hf[i];
    else if (i < 2 * CH * HID) v = ht[i - CH * HID];
    else                       v = hd[i - 2 * CH * HID];
    a0 = fmaf(v, oW[2 * i], a0);
    a1 = fmaf(v, oW[2 * i + 1], a1);
  }
  r0[threadIdx.x] = a0; r1[threadIdx.x] = a1;
  __syncthreads();
  for (int st = 128; st > 0; st >>= 1) {
    if (threadIdx.x < st) {
      r0[threadIdx.x] += r0[threadIdx.x + st];
      r1[threadIdx.x] += r1[threadIdx.x + st];
    }
    __syncthreads();
  }
  if (threadIdx.x == 0) {
    part[blockIdx.x * 2]     = r0[0];
    part[blockIdx.x * 2 + 1] = r1[0];
  }
}

// ---------------- stage 2: reduce partials, bias, sigmoid -> d_out[4]
__global__ void final_reduce(const float* __restrict__ part, const float* __restrict__ ob,
                             float* __restrict__ out, int G) {
  __shared__ float r0[128], r1[128];
  float a0 = 0.f, a1 = 0.f;
  for (int i = threadIdx.x; i < G; i += 128) {
    a0 += part[2 * i];
    a1 += part[2 * i + 1];
  }
  r0[threadIdx.x] = a0; r1[threadIdx.x] = a1;
  __syncthreads();
  for (int st = 64; st > 0; st >>= 1) {
    if (threadIdx.x < st) {
      r0[threadIdx.x] += r0[threadIdx.x + st];
      r1[threadIdx.x] += r1[threadIdx.x + st];
    }
    __syncthreads();
  }
  if (threadIdx.x == 0) {
    float l0 = r0[0] + ob[0], l1 = r1[0] + ob[1];
    out[0] = l0;
    out[1] = l1;
    out[2] = 1.f / (1.f + expf(-l0));
    out[3] = 1.f / (1.f + expf(-l1));
  }
}

extern "C" void kernel_launch(void* const* d_in, const int* in_sizes, int n_in,
                              void* d_out, int out_size, void* d_ws, size_t ws_size,
                              hipStream_t stream) {
  (void)n_in; (void)out_size; (void)ws_size;
  const float* xf   = (const float*)d_in[0];
  const float* xt   = (const float*)d_in[1];
  const float* demo = (const float*)d_in[2];
  const int*   ei   = (const int*)d_in[3];
  const float* W1  = (const float*)d_in[4];
  const float* as1 = (const float*)d_in[5];
  const float* ad1 = (const float*)d_in[6];
  const float* b1  = (const float*)d_in[7];
  const float* W2  = (const float*)d_in[8];
  const float* as2 = (const float*)d_in[9];
  const float* ad2 = (const float*)d_in[10];
  const float* b2  = (const float*)d_in[11];
  const float* W3  = (const float*)d_in[12];
  const float* as3 = (const float*)d_in[13];
  const float* ad3 = (const float*)d_in[14];
  const float* b3  = (const float*)d_in[15];
  const float* W4  = (const float*)d_in[16];
  const float* as4 = (const float*)d_in[17];
  const float* ad4 = (const float*)d_in[18];
  const float* b4  = (const float*)d_in[19];
  const float* fcW = (const float*)d_in[20];
  const float* fcb = (const float*)d_in[21];
  const float* oW  = (const float*)d_in[22];
  const float* ob  = (const float*)d_in[23];

  const int K1 = in_sizes[0] / CH;   // 1280
  const int K3 = in_sizes[1] / CH;   // 4096
  const int E  = in_sizes[3] / 2;    // 3844
  const int Etot = E + CH;

  // workspace carve-up (256B aligned)
  char* ws = (char*)d_ws;
  size_t off = 0;
  auto take = [&](size_t bytes) {
    char* p = ws + off;
    off = (off + bytes + 255) & ~(size_t)255;
    return p;
  };
  u16*   Apack = (u16*)  take((size_t)64 * HID * 2);  // bf16 A matrix, padded to M=64
  float* G     = (float*)take((size_t)64 * HID * 4);  // GEMM output (padded rows unused)
  float* HF    = (float*)take((size_t)CH * HID * 4);  // frequency branch final
  float* HT    = (float*)take((size_t)CH * HID * 4);  // time branch final
  float* S     = (float*)take(64 * 4);
  float* D     = (float*)take(64 * 4);
  int*   OFFS  = (int*)  take(64 * 4);
  int*   SRC   = (int*)  take((size_t)Etot * 4);
  float* ALPHA = (float*)take((size_t)Etot * 4);
  float* HD    = (float*)take((size_t)HID * 4);
  float* PART  = (float*)take(128 * 2 * 4);

  const dim3 aggGrid(HID / 256, CH);
  const int  FINAL_BLOCKS = 128;

  build_csr<<<1, 256, 0, stream>>>(ei, E, OFFS, SRC);

  // ---- GAT1 (freq):  h1 = relu(gat(xf, W1))
  pack_bf16<<<(CH * K1 + 255) / 256, 256, 0, stream>>>(xf, Apack, CH * K1);
  gat_gemm<<<HID / 64, 128, 0, stream>>>((const __bf16*)Apack, W1, G, K1);
  row_dots<<<CH, 256, 0, stream>>>(G, as1, ad1, S, D, HID);
  edge_softmax<<<CH, 64, 0, stream>>>(S, D, OFFS, SRC, ALPHA);
  aggregate<<<aggGrid, 256, 0, stream>>>(G, ALPHA, OFFS, SRC, b1, HF, Apack, HID, 1);

  // ---- GAT2: hf = gat(h1, W2)
  gat_gemm<<<HID / 64, 128, 0, stream>>>((const __bf16*)Apack, W2, G, HID);
  row_dots<<<CH, 256, 0, stream>>>(G, as2, ad2, S, D, HID);
  edge_softmax<<<CH, 64, 0, stream>>>(S, D, OFFS, SRC, ALPHA);
  aggregate<<<aggGrid, 256, 0, stream>>>(G, ALPHA, OFFS, SRC, b2, HF, Apack, HID, 0);

  // ---- GAT3 (time): h3 = relu(gat(xt, W3))
  pack_bf16<<<(CH * K3 + 255) / 256, 256, 0, stream>>>(xt, Apack, CH * K3);
  gat_gemm<<<HID / 64, 128, 0, stream>>>((const __bf16*)Apack, W3, G, K3);
  row_dots<<<CH, 256, 0, stream>>>(G, as3, ad3, S, D, HID);
  edge_softmax<<<CH, 64, 0, stream>>>(S, D, OFFS, SRC, ALPHA);
  aggregate<<<aggGrid, 256, 0, stream>>>(G, ALPHA, OFFS, SRC, b3, HT, Apack, HID, 1);

  // ---- GAT4: ht = gat(h3, W4)
  gat_gemm<<<HID / 64, 128, 0, stream>>>((const __bf16*)Apack, W4, G, HID);
  row_dots<<<CH, 256, 0, stream>>>(G, as4, ad4, S, D, HID);
  edge_softmax<<<CH, 64, 0, stream>>>(S, D, OFFS, SRC, ALPHA);
  aggregate<<<aggGrid, 256, 0, stream>>>(G, ALPHA, OFFS, SRC, b4, HT, Apack, HID, 0);

  // ---- demographics branch + final projection
  fc_demo<<<HID / 256, 256, 0, stream>>>(demo, fcW, fcb, HD);
  final_partial<<<FINAL_BLOCKS, 256, 0, stream>>>(HF, HT, HD, oW, PART,
                                                  (2 * CH + 1) * HID);
  final_reduce<<<1, 128, 0, stream>>>(PART, ob, (float*)d_out, FINAL_BLOCKS);
}
